// CrystalToxicityTransformer_88244398064310
// MI455X (gfx1250) — matrix operand order repaired and checked by reading, not compiled, not run on hardware
//
#include <hip/hip_runtime.h>
#include <hip/hip_bf16.h>
#include <cstdint>
#include <cstddef>

// ---------------------------------------------------------------------------
// CDNA5 (gfx1250) implementation. wave32, WMMA 16x16x32 bf16 with f32 accum.
// All GEMM K dims are multiples of 32 (layer-0 din=16 is zero-padded to 32),
// so fragment loaders are branch-free b128 loads; B is pre-packed to the
// WMMA fragment layout so each lane reads 32 contiguous bytes.
// ---------------------------------------------------------------------------

typedef __bf16 bf16t;
typedef __attribute__((ext_vector_type(16))) __bf16 v16bf;
typedef __attribute__((ext_vector_type(8)))  __bf16 v8bf;
typedef __attribute__((ext_vector_type(8)))  float  v8f;

static constexpr int kNodes  = 50000;
static constexpr int kEdges  = 300000;
static constexpr int kGraphs = 128;
static constexpr int kD      = 256;
#define LN_EPS 1e-5f

// ------------------------------ device utils ------------------------------

__device__ __forceinline__ float wred32(float v) {
#pragma unroll
  for (int m = 16; m > 0; m >>= 1) v += __shfl_xor(v, m, 32);
  return v;
}

__device__ __forceinline__ void atomicMaxF(float* a, float v) {
  if (v >= 0.f) atomicMax((int*)a, __float_as_int(v));
  else          atomicMin((unsigned int*)a, (unsigned int)__float_as_int(v));
}

__device__ __forceinline__ float leaky1(float x) { return x >= 0.f ? x : 0.1f * x; }

// A fragment (ISA 7.12.2): lane half hf covers K runs [8hf,8hf+8) and
// [16+8hf,16+8hf+8) -> two contiguous 16B loads.
__device__ __forceinline__ void load_a(const bf16t* A, int K, int row, int kb,
                                       int hf, v16bf& a) {
  const bf16t* p = A + (size_t)row * K + kb + 8 * hf;
  v8bf lo = *(const v8bf*)p;
  v8bf hi = *(const v8bf*)(p + 16);
  a = __builtin_shufflevector(lo, hi, 0, 1, 2, 3, 4, 5, 6, 7,
                              8, 9, 10, 11, 12, 13, 14, 15);
}

__device__ __forceinline__ void load_a(const float* A, int K, int row, int kb,
                                       int hf, v16bf& a) {
  const float* p = A + (size_t)row * K + kb + 8 * hf;
  float4 x0 = *(const float4*)p;
  float4 x1 = *(const float4*)(p + 4);
  float4 y0 = *(const float4*)(p + 16);
  float4 y1 = *(const float4*)(p + 20);
  a[0] = (bf16t)x0.x; a[1] = (bf16t)x0.y; a[2]  = (bf16t)x0.z; a[3]  = (bf16t)x0.w;
  a[4] = (bf16t)x1.x; a[5] = (bf16t)x1.y; a[6]  = (bf16t)x1.z; a[7]  = (bf16t)x1.w;
  a[8] = (bf16t)y0.x; a[9] = (bf16t)y0.y; a[10] = (bf16t)y0.z; a[11] = (bf16t)y0.w;
  a[12]= (bf16t)y1.x; a[13]= (bf16t)y1.y; a[14] = (bf16t)y1.z; a[15] = (bf16t)y1.w;
}

// B pre-packed fragment-major: [K/32][N/16][lane(32)][16 bf16] -> one 32B read.
__device__ __forceinline__ void load_b_packed(const bf16t* Bp, int N, int tn,
                                              int kb, int lane, v16bf& b) {
  const bf16t* p = Bp + ((size_t)(kb >> 5) * (N >> 4) + tn) * 512 + lane * 16;
  v8bf lo = *(const v8bf*)p;
  v8bf hi = *(const v8bf*)(p + 8);
  b = __builtin_shufflevector(lo, hi, 0, 1, 2, 3, 4, 5, 6, 7,
                              8, 9, 10, 11, 12, 13, 14, 15);
}

__device__ __forceinline__ v8f wmma_bf16(v16bf a, v16bf b, v8f c) {
  return __builtin_amdgcn_wmma_f32_16x16x32_bf16(false, a, false, b, (short)0, c,
                                                 false, false);
}

// ------------------------------ small kernels ------------------------------

__global__ void k_zero(float* p, size_t n) {
  size_t i = (size_t)blockIdx.x * 256 + threadIdx.x;
  if (i < n) p[i] = 0.f;
}
__global__ void k_fill(float* p, size_t n, float v) {
  size_t i = (size_t)blockIdx.x * 256 + threadIdx.x;
  if (i < n) p[i] = v;
}

// z_bf[r, 0:dpad] = bf16(h[r,0:din] + agg[r,0:din]), zero-padded to dpad cols
__global__ void k_add_bf_pad(const float* __restrict__ a, const float* __restrict__ b,
                             bf16t* __restrict__ o, int rows, int din, int dpad) {
  int i = blockIdx.x * 256 + threadIdx.x;
  if (i >= rows * dpad) return;
  int r = i / dpad, c = i - r * dpad;
  o[i] = (bf16t)((c < din) ? (a[(size_t)r * din + c] + b[(size_t)r * din + c]) : 0.f);
}

// Pack weight f32[Ksrc,N] -> bf16 fragment-major [Kpad/32][N/16][lane][16]
// lane covers column tn*16+(lane&15), K runs [kblk*32+8*(lane>>4), +8) and [kblk*32+16+8*(lane>>4), +8).
__global__ void k_pack_b(const float* __restrict__ B, bf16t* __restrict__ out,
                         int Ksrc, int Kpad, int N) {
  int idx = blockIdx.x * 256 + threadIdx.x;
  int total = (Kpad / 32) * (N / 16) * 32;
  if (idx >= total) return;
  int lane = idx & 31, tile = idx >> 5;
  int tn = tile % (N / 16), kblk = tile / (N / 16);
  int hf = lane >> 4, col = tn * 16 + (lane & 15);
  int kbase = kblk * 32 + 8 * hf;
  bf16t vals[16];
#pragma unroll
  for (int e = 0; e < 16; ++e) {
    int k = kbase + (e < 8 ? e : e + 8);
    vals[e] = (bf16t)((k < Ksrc) ? B[(size_t)k * N + col] : 0.f);
  }
  __builtin_memcpy(out + (size_t)tile * 512 + lane * 16, vals, 32);
}

// ------------------------- edge encoder (rank-1+LN) ------------------------
__global__ void k_edge_enc(const float* __restrict__ ea, const float* __restrict__ w,
                           const float* __restrict__ b, const float* __restrict__ g,
                           const float* __restrict__ lb, bf16t* __restrict__ out,
                           int nE) {
  int wave = threadIdx.x >> 5, lane = threadIdx.x & 31;
  int e = blockIdx.x * 8 + wave;
  if (e >= nE) return;
  float a = ea[e];
  float vals[8], s = 0.f, s2 = 0.f;
#pragma unroll
  for (int j = 0; j < 8; ++j) {
    int c = lane * 8 + j;
    float x = leaky1(a * w[c] + b[c]);
    vals[j] = x; s += x; s2 += x * x;
  }
  s = wred32(s); s2 = wred32(s2);
  float mu = s * (1.f / 256.f);
  float rs = rsqrtf(s2 * (1.f / 256.f) - mu * mu + LN_EPS);
  bf16t o[8];
#pragma unroll
  for (int j = 0; j < 8; ++j) {
    int c = lane * 8 + j;
    o[j] = (bf16t)(g[c] * (vals[j] - mu) * rs + lb[c]);
  }
  uint4 pack; __builtin_memcpy(&pack, o, 16);
  *(uint4*)(out + (size_t)e * 256 + lane * 8) = pack;
}

// ------------------------------- WMMA GEMMs --------------------------------
// C = act(A@Bp + bias). Block = 4 waves along N; each wave one 16x16 tile.
template <typename TA>
__global__ void k_gemm(const TA* __restrict__ A, const bf16t* __restrict__ Bp,
                       const float* __restrict__ bias, float* __restrict__ C,
                       bf16t* __restrict__ CB, int M, int N, int K, int act) {
  int wave = threadIdx.x >> 5, lane = threadIdx.x & 31;
  int hf = lane >> 4, r = lane & 15;
  int tn = blockIdx.x * 4 + wave, tm = blockIdx.y;
  if (tn * 16 >= N) return;
  int row = tm * 16 + r, col = tn * 16 + r;
  v8f acc = {};
  for (int kb = 0; kb < K; kb += 32) {
    v16bf a, b;
    load_a(A, K, row, kb, hf, a);
    load_b_packed(Bp, N, tn, kb, lane, b);
    acc = wmma_bf16(a, b, acc);
  }
#pragma unroll
  for (int v = 0; v < 8; ++v) {
    int orow = tm * 16 + 8 * hf + v;
    float x = acc[v] + (bias ? bias[col] : 0.f);
    if (act == 1) x = leaky1(x);
    else if (act == 2) x = x > 0.f ? x : 0.f;
    if (C)  C[(size_t)orow * N + col] = x;
    if (CB) CB[(size_t)orow * N + col] = (bf16t)x;
  }
}

// GEMM + bias + (act) + (residual) + row-LayerNorm. One block owns a full
// 16-row x N slab (N/16 waves), stages it in LDS, LN via wave32 shuffles.
template <typename TA>
__global__ void k_gemm_ln(const TA* __restrict__ A, const bf16t* __restrict__ Bp,
                          const float* __restrict__ bias,
                          const float* __restrict__ resid,
                          const float* __restrict__ g, const float* __restrict__ lb,
                          float* __restrict__ C, bf16t* __restrict__ CB,
                          int M, int N, int K, int act) {
  extern __shared__ float smem[];  // [16][N]
  int wave = threadIdx.x >> 5, lane = threadIdx.x & 31;
  int hf = lane >> 4, r = lane & 15;
  int tm = blockIdx.x, tn = wave;
  {
    int row = tm * 16 + r, col = tn * 16 + r;
    v8f acc = {};
    for (int kb = 0; kb < K; kb += 32) {
      v16bf a, b;
      load_a(A, K, row, kb, hf, a);
      load_b_packed(Bp, N, tn, kb, lane, b);
      acc = wmma_bf16(a, b, acc);
    }
#pragma unroll
    for (int v = 0; v < 8; ++v) {
      int lrow = 8 * hf + v;
      float x = acc[v] + (bias ? bias[col] : 0.f);
      if (act == 1) x = leaky1(x);
      if (resid) x += resid[(size_t)(tm * 16 + lrow) * N + col];
      smem[lrow * N + col] = x;
    }
  }
  __syncthreads();
  if (wave < 16) {
    int rrow = wave, per = N / 32;
    float s = 0.f, s2 = 0.f;
    for (int i = 0; i < per; ++i) {
      float x = smem[rrow * N + lane * per + i];
      s += x; s2 += x * x;
    }
    s = wred32(s); s2 = wred32(s2);
    float mu = s / (float)N;
    float rs = rsqrtf(s2 / (float)N - mu * mu + LN_EPS);
    size_t grow = (size_t)(tm * 16 + rrow) * N;
    for (int i = 0; i < per; ++i) {
      int c = lane * per + i;
      float y = g[c] * (smem[rrow * N + c] - mu) * rs + lb[c];
      if (C)  C[grow + c] = y;
      if (CB) CB[grow + c] = (bf16t)y;
    }
  }
}

// Fused GINE edge projection + message + scatter:
// e2 = e@lin_w + lin_b; msg = relu(h[src]+e2); atomicAdd(agg[dst], msg)
__global__ void k_gine_msg(const bf16t* __restrict__ Ebf, const bf16t* __restrict__ Wp,
                           const float* __restrict__ lb, const float* __restrict__ H,
                           const int* __restrict__ src, const int* __restrict__ dst,
                           float* __restrict__ Agg, int nE, int din, int K) {
  int wave = threadIdx.x >> 5, lane = threadIdx.x & 31;
  int hf = lane >> 4, r = lane & 15;
  int tn = blockIdx.x * 4 + wave, tm = blockIdx.y;
  if (tn * 16 >= din) return;
  int row = tm * 16 + r, col = tn * 16 + r;
  v8f acc = {};
  for (int kb = 0; kb < K; kb += 32) {
    v16bf a, b;
    load_a(Ebf, K, row, kb, hf, a);
    load_b_packed(Wp, din, tn, kb, lane, b);
    acc = wmma_bf16(a, b, acc);
  }
#pragma unroll
  for (int v = 0; v < 8; ++v) {
    int edge = tm * 16 + 8 * hf + v;
    int s = src[edge], d = dst[edge];
    float msg = acc[v] + lb[col] + H[(size_t)s * din + col];
    if (msg > 0.f) atomicAdd(&Agg[(size_t)d * din + col], msg);
  }
}

// ------------------------------- BatchNorm ---------------------------------
__global__ void k_bn_stats(const float* __restrict__ X, const float* __restrict__ bias,
                           float* __restrict__ stats, int Nn) {
  int c = threadIdx.x, r0 = blockIdx.x * 64;
  float s = 0.f, s2 = 0.f;
  for (int i = 0; i < 64; ++i) {
    int r = r0 + i;
    if (r < Nn) {
      float x = X[(size_t)r * 256 + c] + (bias ? bias[c] : 0.f);
      s += x; s2 += x * x;
    }
  }
  atomicAdd(&stats[c], s);
  atomicAdd(&stats[256 + c], s2);
}

__global__ void k_bn_apply(const float* __restrict__ X, const float* __restrict__ bias,
                           const float* __restrict__ stats, const float* __restrict__ g,
                           const float* __restrict__ b, float* __restrict__ Y,
                           bf16t* __restrict__ Yb, int Nn) {
  size_t idx = (size_t)blockIdx.x * 256 + threadIdx.x;
  if (idx >= (size_t)Nn * 256) return;
  int c = idx & 255;
  float mu = stats[c] / (float)Nn;
  float var = stats[256 + c] / (float)Nn - mu * mu;
  float x = X[idx] + (bias ? bias[c] : 0.f);
  float y = leaky1(g[c] * (x - mu) * rsqrtf(var + LN_EPS) + b[c]);
  Y[idx] = y;
  if (Yb) Yb[idx] = (bf16t)y;
}

// --------------------------------- GAT -------------------------------------
__global__ void k_gat_coef(const float* __restrict__ xt, const float* __restrict__ as,
                           const float* __restrict__ ad, float* __restrict__ As,
                           float* __restrict__ Ad, int Nn) {
  int wave = threadIdx.x >> 5, lane = threadIdx.x & 31;
  int node = blockIdx.x * 8 + wave;
  if (node >= Nn) return;
  float ss = 0.f, sd = 0.f;
#pragma unroll
  for (int j = 0; j < 8; ++j) {
    int c = lane * 8 + j, hd = c >> 6, dim = c & 63;
    float x = xt[(size_t)node * 256 + c];
    ss += x * as[hd * 64 + dim];
    sd += x * ad[hd * 64 + dim];
  }
#pragma unroll
  for (int m = 1; m < 8; m <<= 1) { ss += __shfl_xor(ss, m, 32); sd += __shfl_xor(sd, m, 32); }
  if ((lane & 7) == 0) {
    int hd = lane >> 3;
    As[node * 4 + hd] = ss;
    Ad[node * 4 + hd] = sd;
  }
}

__global__ void k_gat_logits(const float* __restrict__ As, const float* __restrict__ Ad,
                             const int* __restrict__ src, const int* __restrict__ dst,
                             float* __restrict__ L, float* __restrict__ Mx,
                             int nE, int E2) {
  int e = blockIdx.x * 256 + threadIdx.x;
  if (e >= E2) return;
  int s, d;
  if (e < nE) { s = src[e]; d = dst[e]; } else { s = d = e - nE; }
#pragma unroll
  for (int h = 0; h < 4; ++h) {
    float l = As[s * 4 + h] + Ad[d * 4 + h];
    l = l >= 0.f ? l : 0.2f * l;
    L[(size_t)e * 4 + h] = l;
    atomicMaxF(&Mx[d * 4 + h], l);
  }
}

__global__ void k_gat_expsum(float* __restrict__ L, const float* __restrict__ Mx,
                             const int* __restrict__ src, const int* __restrict__ dst,
                             float* __restrict__ Den, int nE, int E2) {
  int e = blockIdx.x * 256 + threadIdx.x;
  if (e >= E2) return;
  int d = (e < nE) ? dst[e] : e - nE;
#pragma unroll
  for (int h = 0; h < 4; ++h) {
    float ee = __expf(L[(size_t)e * 4 + h] - Mx[d * 4 + h]);
    L[(size_t)e * 4 + h] = ee;
    atomicAdd(&Den[d * 4 + h], ee);
  }
}

__global__ void k_gat_scatter(const float* __restrict__ L, const float* __restrict__ Den,
                              const float* __restrict__ xt, const int* __restrict__ src,
                              const int* __restrict__ dst, float* __restrict__ Out,
                              int nE, int E2) {
  int t = threadIdx.x;
  int e = blockIdx.x * 4 + (t >> 6);
  if (e >= E2) return;
  int s, d;
  if (e < nE) { s = src[e]; d = dst[e]; } else { s = d = e - nE; }
  int c0 = (t & 63) * 4, h = c0 >> 6;
  float alpha = L[(size_t)e * 4 + h] / Den[d * 4 + h];
#pragma unroll
  for (int j = 0; j < 4; ++j) {
    atomicAdd(&Out[(size_t)d * 256 + c0 + j], alpha * xt[(size_t)s * 256 + c0 + j]);
  }
}

// -------------------------------- pooling ----------------------------------
__global__ void k_pool(const float* __restrict__ X, const int* __restrict__ batch,
                       float* __restrict__ gsum, float* __restrict__ cnt, int Nn) {
  int c = threadIdx.x, r0 = blockIdx.x * 64;
  for (int i = 0; i < 64; ++i) {
    int r = r0 + i;
    if (r < Nn) {
      int g = batch[r];
      atomicAdd(&gsum[(size_t)g * 256 + c], X[(size_t)r * 256 + c]);
      if (c == 0) atomicAdd(&cnt[g], 1.f);
    }
  }
}
__global__ void k_pool_div(const float* __restrict__ gsum, const float* __restrict__ cnt,
                           float* __restrict__ gemb) {
  int idx = blockIdx.x * 256 + threadIdx.x;
  gemb[idx] = gsum[idx] / fmaxf(cnt[idx >> 8], 1.f);
}

__global__ void k_addpos(float* __restrict__ t, const float* __restrict__ pos, int n) {
  int i = blockIdx.x * 256 + threadIdx.x;
  if (i < n) t[i] += pos[i & 255];  // pos_emb row 0
}

__global__ void k_concat_bf(const float* __restrict__ ge, const float* __restrict__ t,
                            bf16t* __restrict__ out, int nG) {
  int i = blockIdx.x * 256 + threadIdx.x;
  if (i >= nG * 512) return;
  int g = i >> 9, c = i & 511;
  out[i] = (bf16t)(c < 256 ? ge[g * 256 + c] : t[g * 256 + c - 256]);
}

__global__ void k_pred(const float* __restrict__ y1, const float* __restrict__ w2,
                       const float* __restrict__ b2, float* __restrict__ out, int nG) {
  int wave = threadIdx.x >> 5, lane = threadIdx.x & 31;
  int g = blockIdx.x * 8 + wave;
  if (g >= nG) return;
  float s = 0.f;
#pragma unroll
  for (int j = 0; j < 8; ++j) {
    int c = lane * 8 + j;
    s += y1[(size_t)g * 256 + c] * w2[c];
  }
  s = wred32(s);
  if (lane == 0) out[g] = s + b2[0];
}

// ------------------------------- host side ---------------------------------

extern "C" void kernel_launch(void* const* d_in, const int* in_sizes, int n_in,
                              void* d_out, int out_size, void* d_ws, size_t ws_size,
                              hipStream_t stream) {
  (void)in_sizes; (void)n_in; (void)out_size; (void)ws_size;
  auto F = [&](int i) { return (const float*)d_in[i]; };
  // params (jax pytree flatten order: sorted dict keys), then data inputs
  const float* x_in    = F(103);
  const float* eattr   = F(104);
  const float* expfeat = F(105);
  const int*   ei      = (const int*)d_in[106];
  const int*   batchv  = (const int*)d_in[107];
  const int *src = ei, *dst = ei + kEdges;
  float* outp = (float*)d_out;

  // ---- workspace bump allocator ----
  char* ws = (char*)d_ws; size_t off = 0;
  auto WA = [&](size_t bytes) { char* p = ws + off; off = (off + bytes + 255) & ~(size_t)255; return p; };
  bf16t* e_bf  = (bf16t*)WA((size_t)kEdges * kD * 2);    // edge embeddings (bf16: fits L2)
  float* bufA  = (float*)WA((size_t)kNodes * kD * 4);    // h / gat-out
  float* agg   = (float*)WA((size_t)kNodes * kD * 4);    // agg / z3 / xt
  bf16t* hbf   = (bf16t*)WA((size_t)kNodes * kD * 2);    // z_bf / h_bf
  bf16t* zbf2  = (bf16t*)WA((size_t)kNodes * kD * 2);    // z2_bf
  float* stats = (float*)WA(512 * 4);
  float* As_   = (float*)WA((size_t)kNodes * 4 * 4);
  float* Ad_   = (float*)WA((size_t)kNodes * 4 * 4);
  float* Mx    = (float*)WA((size_t)kNodes * 4 * 4);
  float* Den   = (float*)WA((size_t)kNodes * 4 * 4);
  float* Lbuf  = (float*)WA((size_t)(kEdges + kNodes) * 4 * 4);
  float* gsum  = (float*)WA((size_t)kGraphs * kD * 4);
  float* cnt   = (float*)WA(kGraphs * 4);
  float* gemb  = (float*)WA((size_t)kGraphs * kD * 4);
  float* tbuf  = (float*)WA((size_t)kGraphs * kD * 4);
  bf16t* vbf   = (bf16t*)WA((size_t)kGraphs * kD * 2);
  bf16t* f1bf  = (bf16t*)WA((size_t)kGraphs * 1024 * 2);
  bf16t* catbf = (bf16t*)WA((size_t)kGraphs * 512 * 2);
  bf16t* fzbf  = (bf16t*)WA((size_t)kGraphs * 512 * 2);
  float* y1    = (float*)WA((size_t)kGraphs * kD * 4);

  // Pack a weight f32[Ksrc,N] into bf16 WMMA-fragment layout (K padded).
  auto packB = [&](const float* w, int Ksrc, int Kpad, int N) {
    size_t total = (size_t)(Kpad / 32) * (N / 16) * 512;
    bf16t* p = (bf16t*)WA(total * 2);
    int threads = (Kpad / 32) * (N / 16) * 32;
    k_pack_b<<<(threads + 255) / 256, 256, 0, stream>>>(w, p, Ksrc, Kpad, N);
    return p;
  };
  auto zero = [&](float* p, size_t n) {
    k_zero<<<(unsigned)((n + 255) / 256), 256, 0, stream>>>(p, n);
  };

  // ---- edge encoder ----
  k_edge_enc<<<kEdges / 8, 256, 0, stream>>>(eattr, F(3), F(0), F(2), F(1), e_bf, kEdges);

  // ---- 3x GINEConv ----
  int dins[3] = {16, kD, kD};
  const float* Hcur = x_in;
  for (int l = 0; l < 3; ++l) {
    int gb = 18 + 10 * l, din = dins[l];
    int dpad = (din < 32) ? 32 : din;  // K padding for the z@w1 GEMM
    const float *b1 = F(gb + 0), *b2 = F(gb + 1), *bn_b = F(gb + 2), *bn_g = F(gb + 3);
    const float *lin_b = F(gb + 4), *lin_w = F(gb + 5);
    const float *ln_b = F(gb + 6), *ln_g = F(gb + 7), *w1 = F(gb + 8), *w2 = F(gb + 9);
    bf16t* linw_p = packB(lin_w, kD, kD, din);
    bf16t* w1_p   = packB(w1, din, dpad, kD);
    bf16t* w2_p   = packB(w2, kD, kD, kD);

    zero(agg, (size_t)kNodes * din);
    k_gine_msg<<<dim3((din + 63) / 64, kEdges / 16), 128, 0, stream>>>(
        e_bf, linw_p, lin_b, Hcur, src, dst, agg, kEdges, din, kD);
    k_add_bf_pad<<<(kNodes * dpad + 255) / 256, 256, 0, stream>>>(
        Hcur, agg, hbf, kNodes, din, dpad);
    k_gemm_ln<bf16t><<<dim3(kNodes / 16), (kD / 16) * 32, 16 * kD * 4, stream>>>(
        hbf, w1_p, b1, nullptr, ln_g, ln_b, nullptr, zbf2, kNodes, kD, dpad, 1);
    k_gemm<bf16t><<<dim3(kD / 64, kNodes / 16), 128, 0, stream>>>(
        zbf2, w2_p, b2, agg, nullptr, kNodes, kD, kD, 0);
    zero(stats, 512);
    k_bn_stats<<<(kNodes + 63) / 64, 256, 0, stream>>>(agg, nullptr, stats, kNodes);
    k_bn_apply<<<(unsigned)(((size_t)kNodes * kD + 255) / 256), 256, 0, stream>>>(
        agg, nullptr, stats, bn_g, bn_b, bufA, hbf, kNodes);
    Hcur = bufA;
  }

  // ---- GAT (4 heads, self loops) ----
  {
    const float *att_d = F(12), *att_s = F(13), *gbias = F(14);
    const float *gbn_b = F(15), *gbn_g = F(16), *gw = F(17);
    bf16t* gw_p = packB(gw, kD, kD, kD);
    k_gemm<bf16t><<<dim3(4, kNodes / 16), 128, 0, stream>>>(
        hbf, gw_p, nullptr, agg, nullptr, kNodes, kD, kD, 0);  // xt -> agg
    k_gat_coef<<<(kNodes + 7) / 8, 256, 0, stream>>>(agg, att_s, att_d, As_, Ad_, kNodes);
    int E2 = kEdges + kNodes;
    k_fill<<<(kNodes * 4 + 255) / 256, 256, 0, stream>>>(Mx, (size_t)kNodes * 4, -1e30f);
    k_gat_logits<<<(E2 + 255) / 256, 256, 0, stream>>>(As_, Ad_, src, dst, Lbuf, Mx, kEdges, E2);
    zero(Den, (size_t)kNodes * 4);
    k_gat_expsum<<<(E2 + 255) / 256, 256, 0, stream>>>(Lbuf, Mx, src, dst, Den, kEdges, E2);
    zero(bufA, (size_t)kNodes * kD);
    k_gat_scatter<<<(E2 + 3) / 4, 256, 0, stream>>>(Lbuf, Den, agg, src, dst, bufA, kEdges, E2);
    zero(stats, 512);
    k_bn_stats<<<(kNodes + 63) / 64, 256, 0, stream>>>(bufA, gbias, stats, kNodes);
    k_bn_apply<<<(unsigned)(((size_t)kNodes * kD + 255) / 256), 256, 0, stream>>>(
        bufA, gbias, stats, gbn_g, gbn_b, bufA, nullptr, kNodes);
  }

  // ---- global mean pool ----
  zero(gsum, (size_t)kGraphs * kD);
  zero(cnt, kGraphs);
  k_pool<<<(kNodes + 63) / 64, 256, 0, stream>>>(bufA, batchv, gsum, cnt, kNodes);
  k_pool_div<<<kGraphs, 256, 0, stream>>>(gsum, cnt, gemb);

  // ---- experimental branch (seq_len==1 => attn==1 => o==V; skip q/k) ----
  {
    bf16t* epw_p = packB(F(7), 64, 64, kD);
    k_gemm_ln<float><<<dim3(kGraphs / 16), (kD / 16) * 32, 16 * kD * 4, stream>>>(
        expfeat, epw_p, F(4), nullptr, F(6), F(5), tbuf, nullptr, kGraphs, kD, 64, 1);
    k_addpos<<<(kGraphs * kD + 255) / 256, 256, 0, stream>>>(tbuf, F(48), kGraphs * kD);
    for (int l = 0; l < 3; ++l) {
      int tb = 55 + 16 * l;
      const float *bo = F(tb + 1), *bv = F(tb + 3);
      const float *ffb1 = F(tb + 4), *ffb2 = F(tb + 5), *ffw1 = F(tb + 6), *ffw2 = F(tb + 7);
      const float *ln1b = F(tb + 8), *ln1g = F(tb + 9), *ln2b = F(tb + 10), *ln2g = F(tb + 11);
      const float *wo = F(tb + 13), *wv = F(tb + 15);
      bf16t* wv_p  = packB(wv, kD, kD, kD);
      bf16t* wo_p  = packB(wo, kD, kD, kD);
      bf16t* f1w_p = packB(ffw1, kD, kD, 1024);
      bf16t* f2w_p = packB(ffw2, 1024, 1024, kD);
      k_gemm<float><<<dim3(4, kGraphs / 16), 128, 0, stream>>>(
          tbuf, wv_p, bv, nullptr, vbf, kGraphs, kD, kD, 0);
      k_gemm_ln<bf16t><<<dim3(kGraphs / 16), (kD / 16) * 32, 16 * kD * 4, stream>>>(
          vbf, wo_p, bo, tbuf, ln1g, ln1b, tbuf, nullptr, kGraphs, kD, kD, 0);
      k_gemm<float><<<dim3(16, kGraphs / 16), 128, 0, stream>>>(
          tbuf, f1w_p, ffb1, nullptr, f1bf, kGraphs, 1024, kD, 2);
      k_gemm_ln<bf16t><<<dim3(kGraphs / 16), (kD / 16) * 32, 16 * kD * 4, stream>>>(
          f1bf, f2w_p, ffb2, tbuf, ln2g, ln2b, tbuf, nullptr, kGraphs, kD, 1024, 0);
    }
  }

  // ---- fusion + predictor ----
  k_concat_bf<<<(kGraphs * 512 + 255) / 256, 256, 0, stream>>>(gemb, tbuf, catbf, kGraphs);
  bf16t* fw_p = packB(F(11), 512, 512, 512);
  k_gemm_ln<bf16t><<<dim3(kGraphs / 16), (512 / 16) * 32, 16 * 512 * 4, stream>>>(
      catbf, fw_p, F(8), nullptr, F(10), F(9), nullptr, fzbf, kGraphs, 512, 512, 1);
  bf16t* pw1_p = packB(F(53), 512, 512, kD);
  k_gemm_ln<bf16t><<<dim3(kGraphs / 16), (kD / 16) * 32, 16 * kD * 4, stream>>>(
      fzbf, pw1_p, F(49), nullptr, F(52), F(51), y1, nullptr, kGraphs, kD, 512, 1);
  k_pred<<<kGraphs / 8, 256, 0, stream>>>(y1, F(54), F(50), outp, kGraphs);
}
